// SimpleTemporalAttn_38122129719765
// MI455X (gfx1250) — matrix operand order repaired and checked by reading, not compiled, not run on hardware
//
#include <hip/hip_runtime.h>
#include <hip/hip_bf16.h>

typedef _Float16 half_t;
typedef __attribute__((ext_vector_type(16))) _Float16 v16h;
typedef __attribute__((ext_vector_type(8)))  _Float16 v8h;
typedef __attribute__((ext_vector_type(8)))  float    v8f;

#define B_     32
#define L_     1024
#define HID_   64
#define NH_    4
#define HD_    16
#define INTER_ 32
#define NL_    4
#define MAXP_  1024
#define BL_    (B_ * L_)
#define QREL_W 1040

// ---------------------------------------------------------------- helpers

__device__ inline v8h zero8h() {
  v8h v;
#pragma unroll
  for (int i = 0; i < 8; ++i) v[i] = (_Float16)0;
  return v;
}
__device__ inline v8f zero8f() {
  v8f v;
#pragma unroll
  for (int i = 0; i < 8; ++i) v[i] = 0.0f;
  return v;
}
__device__ inline v16h cat8(v8h lo, v8h hi) {
  return __builtin_shufflevector(lo, hi, 0, 1, 2, 3, 4, 5, 6, 7, 8, 9, 10, 11,
                                 12, 13, 14, 15);
}

__device__ inline v8f wmma_f16(v16h a, v16h b, v8f c) {
  // v_wmma_f32_16x16x32_f16: (neg_a, A, neg_b, B, c_mod, C, reuse_a, reuse_b)
  return __builtin_amdgcn_wmma_f32_16x16x32_f16(false, a, false, b, (short)0, c,
                                                false, false);
}

// A-tile 16x32 f16 row-major, ld halves per row (rows 16B aligned).
// ISA layout: lane m / m+16; halves 0..7 = K(hi?8..15:0..7),
// halves 8..15 = K(hi?24..31:16..23)  -> two 128-bit loads per lane.
__device__ inline v16h load_a32(const half_t* __restrict__ p, int ld) {
  int lane = threadIdx.x & 31;
  int m = lane & 15;
  int k0 = (lane >> 4) << 3;
  const half_t* row = p + (size_t)m * ld + k0;
  v8h lo = *(const v8h*)(row);
  v8h hi = *(const v8h*)(row + 16);
  return cat8(lo, hi);
}

// A-tile 16x16 padded to K=32 (upper K half zero): one 128-bit load per lane.
__device__ inline v16h load_a16(const half_t* __restrict__ p, int ld) {
  int lane = threadIdx.x & 31;
  int m = lane & 15;
  int k0 = (lane >> 4) << 3;
  const half_t* row = p + (size_t)m * ld + k0;
  v8h lo = *(const v8h*)(row);
  return cat8(lo, zero8h());
}

// B-tile from pre-swizzled weights: tile base points at 32x16 halves laid out
// [lane][16 halves] -> two 128-bit loads per lane.
__device__ inline v16h load_b_swz(const half_t* __restrict__ tile) {
  int lane = threadIdx.x & 31;
  const half_t* p = tile + (lane << 4);
  v8h lo = *(const v8h*)(p);
  v8h hi = *(const v8h*)(p + 8);
  return cat8(lo, hi);
}

// B-tile where column n = a contiguous 16-half row in memory (K=16 padded):
// lane half h loads K 8h..8h+7 into halves 0..7; halves 8..15 zero.
__device__ inline v16h load_b_rows16(const half_t* __restrict__ base,
                                     size_t row_stride, int lanehi) {
  int lane = threadIdx.x & 31;
  const half_t* p = base + (size_t)(lane & 15) * row_stride + ((lane >> 4) << 3);
  v8h lo = *(const v8h*)(p);
  v8h hi = zero8h();
  (void)lanehi;
  return cat8(lo, hi);
}

// ---------------------------------------------------------------- kernels

__global__ void k_cast_f32_f16(const float* __restrict__ src,
                               half_t* __restrict__ dst, int n) {
  int i = blockIdx.x * blockDim.x + threadIdx.x;
  if (i < n) dst[i] = (half_t)src[i];
}

// Convert + swizzle a [NL][K][N] f32 weight into WMMA B-fragment layout:
// dst[layer][ct][kt][lane][i] with lane = (n&15) + 16*((k>>3)&1), i = (k&7) + 8*((k&31)>>4).
__global__ void k_swizzle_w(const float* __restrict__ src,
                            half_t* __restrict__ dst, int K, int N, int total) {
  int i = blockIdx.x * blockDim.x + threadIdx.x;
  if (i >= total) return;
  int per = K * N;
  int layer = i / per;
  int r = i - layer * per;
  int k = r / N, n = r % N;
  int ct = n >> 4;
  int kt = k >> 5;
  int kin = k & 31;
  int lane = (n & 15) + (((kin >> 3) & 1) << 4);
  int KT = K >> 5;
  int idx = ((((ct * KT + kt) << 5) + lane) << 4) + (kin & 7) + ((kin >> 4) << 3);
  dst[(size_t)layer * per + idx] = (half_t)src[i];
}

__global__ void k_init_x(const float* __restrict__ f, float* __restrict__ x32,
                         half_t* __restrict__ x16, int n) {
  int i = blockIdx.x * blockDim.x + threadIdx.x;
  if (i < n) {
    float v = f[i];
    x32[i] = v;
    x16[i] = (half_t)v;
  }
}

// q/k/v = x @ W + b   (grid: BL/16 x 12; z = mat*4 + colTile)
// q,k stored [B,L,HID]; v stored transposed vT[b,h,d,l] for the P@V tile.
__global__ __launch_bounds__(32)
void k_gemm_qkv(const half_t* __restrict__ x, const half_t* __restrict__ wq,
                const half_t* __restrict__ wk, const half_t* __restrict__ wv,
                const float* __restrict__ bq, const float* __restrict__ bk,
                const float* __restrict__ bv, half_t* __restrict__ q,
                half_t* __restrict__ kk, half_t* __restrict__ vT) {
  int row0 = blockIdx.x * 16;
  int z = blockIdx.y;
  int mat = z >> 2;
  int n0 = (z & 3) * 16;
  const half_t* w = (mat == 0) ? wq : (mat == 1) ? wk : wv;
  const float* bias = (mat == 0) ? bq : (mat == 1) ? bk : bv;
  int ct = n0 >> 4;
  v8f c = zero8f();
  c = wmma_f16(load_a32(x + (size_t)row0 * HID_, HID_),
               load_b_swz(w + ((ct * 2 + 0) << 9)), c);
  c = wmma_f16(load_a32(x + (size_t)row0 * HID_ + 32, HID_),
               load_b_swz(w + ((ct * 2 + 1) << 9)), c);
  int lane = threadIdx.x;
  int n = lane & 15, hi = lane >> 4;
  float bb = bias[n0 + n];
  if (mat < 2) {
    half_t* dst = (mat == 0) ? q : kk;
#pragma unroll
    for (int r = 0; r < 8; ++r)
      dst[(size_t)(row0 + r + 8 * hi) * HID_ + n0 + n] = (half_t)(c[r] + bb);
  } else {
    int h = ct;  // head index
#pragma unroll
    for (int r = 0; r < 8; ++r) {
      int row = row0 + r + 8 * hi;  // global b*L + l
      int b = row >> 10, l = row & (L_ - 1);
      vT[(((size_t)(b * NH_ + h) * HD_) + n) * L_ + l] = (half_t)(c[r] + bb);
    }
  }
}

// Fused flash attention + relative-position, per (b,h,16-row block).
__global__ __launch_bounds__(32)
void k_attention(const half_t* __restrict__ q, const half_t* __restrict__ k,
                 const half_t* __restrict__ vT, const half_t* __restrict__ rel,
                 half_t* __restrict__ ctx) {
  __shared__ half_t qrel[16 * QREL_W];  // 33280 B
  __shared__ half_t ptile[16 * 16];     //   512 B
  int l0 = blockIdx.x * 16;
  int bh = blockIdx.y;
  int b = bh >> 2, h = bh & 3;
  int lane = threadIdx.x;
  int n = lane & 15, hi = lane >> 4;

  const half_t* qbase = q + ((size_t)(b * L_ + l0)) * HID_ + h * HD_;
  v16h aq = load_a16(qbase, HID_);  // 16x16 q tile, K padded to 32

  // qrel[m][jj] = q[l0+m,:] . rel_emb[l0+jj,:]  for jj in [0,1039]  (65 WMMAs)
  for (int jt = 0; jt < 65; ++jt) {
    int j = l0 + jt * 16 + n;
    if (j > 2 * MAXP_ - 2) j = 2 * MAXP_ - 2;
    v8h lo = *(const v8h*)(rel + (size_t)j * HD_ + (hi << 3));
    v16h bt = cat8(lo, zero8h());
    v8f c = wmma_f16(aq, bt, zero8f());
#pragma unroll
    for (int r = 0; r < 8; ++r)
      qrel[(r + 8 * hi) * QREL_W + jt * 16 + n] = (half_t)c[r];
  }
  __syncthreads();

  float mstat[8], lstat[8];
#pragma unroll
  for (int r = 0; r < 8; ++r) {
    mstat[r] = -3.0e38f;
    lstat[r] = 0.0f;
  }
  v8f acc = zero8f();
  const float scale = 0.25f;  // 1/sqrt(HEAD_DIM)

  const half_t* vbase = vT + ((size_t)(b * NH_ + h) * HD_) * L_;

  for (int rt = 0; rt < L_ / 16; ++rt) {
    int r0 = rt * 16;
    // scores tile: A=q, B[d][n]=k[r0+n][d]  (k rows are contiguous 16 halves)
    v16h bk = load_b_rows16(
        k + ((size_t)(b * L_ + r0)) * HID_ + h * HD_, HID_, hi);
    v8f s = wmma_f16(aq, bk, zero8f());

    float sv[8];
#pragma unroll
    for (int r = 0; r < 8; ++r) {
      int m = r + 8 * hi;
      float relv = (float)qrel[m * QREL_W + (m - n + (MAXP_ - 1) - r0)];
      sv[r] = (s[r] + relv) * scale;
    }
    // row max across the 16 lanes holding a row
    float tm[8];
#pragma unroll
    for (int r = 0; r < 8; ++r) tm[r] = sv[r];
    for (int off = 1; off < 16; off <<= 1) {
#pragma unroll
      for (int r = 0; r < 8; ++r)
        tm[r] = fmaxf(tm[r], __shfl_xor(tm[r], off, 32));
    }
    float pv[8], corr[8];
#pragma unroll
    for (int r = 0; r < 8; ++r) {
      float mnew = fmaxf(mstat[r], tm[r]);
      corr[r] = __expf(mstat[r] - mnew);
      pv[r] = __expf(sv[r] - mnew);
      mstat[r] = mnew;
    }
    float ps[8];
#pragma unroll
    for (int r = 0; r < 8; ++r) ps[r] = pv[r];
    for (int off = 1; off < 16; off <<= 1) {
#pragma unroll
      for (int r = 0; r < 8; ++r) ps[r] += __shfl_xor(ps[r], off, 32);
    }
#pragma unroll
    for (int r = 0; r < 8; ++r) lstat[r] = lstat[r] * corr[r] + ps[r];

    // P tile: C-layout -> LDS -> A-layout (transpose within the wave)
#pragma unroll
    for (int r = 0; r < 8; ++r) ptile[(r + 8 * hi) * 16 + n] = (half_t)pv[r];
    __syncthreads();
    v16h ap;
    {
      int m2 = lane & 15;
      int k0 = hi << 3;
      v8h lo = *(const v8h*)(ptile + m2 * 16 + k0);
      ap = cat8(lo, zero8h());
    }
    __syncthreads();
    // B = V tile: B[kr][d] = vT[d][r0+kr] -> contiguous 16 halves per lane
    v16h bvv = load_b_rows16(vbase + r0, L_, hi);
#pragma unroll
    for (int r = 0; r < 8; ++r) acc[r] *= corr[r];
    acc = wmma_f16(ap, bvv, acc);
  }
#pragma unroll
  for (int r = 0; r < 8; ++r) {
    int m = r + 8 * hi;
    ctx[((size_t)(b * L_ + l0 + m)) * HID_ + h * HD_ + n] =
        (half_t)(acc[r] / lstat[r]);
  }
}

// out = a@w + bias + res  (K=64, f32 result), swizzled weights
__global__ __launch_bounds__(32)
void k_proj_res(const half_t* __restrict__ a, const half_t* __restrict__ w,
                const float* __restrict__ bias, const float* __restrict__ res,
                float* __restrict__ outp) {
  int row0 = blockIdx.x * 16;
  int n0 = blockIdx.y * 16;
  int ct = n0 >> 4;
  v8f c = zero8f();
  c = wmma_f16(load_a32(a + (size_t)row0 * HID_, HID_),
               load_b_swz(w + ((ct * 2 + 0) << 9)), c);
  c = wmma_f16(load_a32(a + (size_t)row0 * HID_ + 32, HID_),
               load_b_swz(w + ((ct * 2 + 1) << 9)), c);
  int lane = threadIdx.x;
  int n = lane & 15, hi = lane >> 4;
  float bb = bias[n0 + n];
#pragma unroll
  for (int r = 0; r < 8; ++r) {
    size_t idx = (size_t)(row0 + r + 8 * hi) * HID_ + n0 + n;
    outp[idx] = c[r] + bb + res[idx];
  }
}

// inter = gelu(x@Wi + bi)  (K=64, N=32), swizzled weights
__global__ __launch_bounds__(32)
void k_ffn_up(const half_t* __restrict__ x, const half_t* __restrict__ wi,
              const float* __restrict__ bi, half_t* __restrict__ inter) {
  int row0 = blockIdx.x * 16;
  int n0 = blockIdx.y * 16;
  int ct = n0 >> 4;
  v8f c = zero8f();
  c = wmma_f16(load_a32(x + (size_t)row0 * HID_, HID_),
               load_b_swz(wi + ((ct * 2 + 0) << 9)), c);
  c = wmma_f16(load_a32(x + (size_t)row0 * HID_ + 32, HID_),
               load_b_swz(wi + ((ct * 2 + 1) << 9)), c);
  int lane = threadIdx.x;
  int n = lane & 15, hi = lane >> 4;
  float bb = bi[n0 + n];
#pragma unroll
  for (int r = 0; r < 8; ++r) {
    float u = c[r] + bb;
    float g = 0.5f * u * (1.0f + erff(u * 0.70710678118654752440f));
    inter[(size_t)(row0 + r + 8 * hi) * INTER_ + n0 + n] = (half_t)g;
  }
}

// y = inter@Wd + bd + res  (K=32 exactly -> one WMMA), swizzled weights
__global__ __launch_bounds__(32)
void k_ffn_down(const half_t* __restrict__ inter, const half_t* __restrict__ wd,
                const float* __restrict__ bd, const float* __restrict__ res,
                float* __restrict__ outp) {
  int row0 = blockIdx.x * 16;
  int n0 = blockIdx.y * 16;
  int ct = n0 >> 4;  // KT = 1
  v8f c = zero8f();
  c = wmma_f16(load_a32(inter + (size_t)row0 * INTER_, INTER_),
               load_b_swz(wd + (ct << 9)), c);
  int lane = threadIdx.x;
  int n = lane & 15, hi = lane >> 4;
  float bb = bd[n0 + n];
#pragma unroll
  for (int r = 0; r < 8; ++r) {
    size_t idx = (size_t)(row0 + r + 8 * hi) * HID_ + n0 + n;
    outp[idx] = c[r] + bb + res[idx];
  }
}

// wave-per-row LayerNorm over 64 cols; writes f32 + f16 copies of x
__global__ __launch_bounds__(32)
void k_layernorm(const float* __restrict__ y, const float* __restrict__ g,
                 const float* __restrict__ bta, float* __restrict__ x32,
                 half_t* __restrict__ x16) {
  int row = blockIdx.x;
  int lane = threadIdx.x;
  float a0 = y[(size_t)row * HID_ + lane];
  float a1 = y[(size_t)row * HID_ + 32 + lane];
  float s = a0 + a1;
  for (int off = 1; off < 32; off <<= 1) s += __shfl_xor(s, off, 32);
  float mean = s * (1.0f / 64.0f);
  float d0 = a0 - mean, d1 = a1 - mean;
  float vs = d0 * d0 + d1 * d1;
  for (int off = 1; off < 32; off <<= 1) vs += __shfl_xor(vs, off, 32);
  float inv = rsqrtf(vs * (1.0f / 64.0f) + 1e-12f);
  float o0 = d0 * inv * g[lane] + bta[lane];
  float o1 = d1 * inv * g[32 + lane] + bta[32 + lane];
  x32[(size_t)row * HID_ + lane] = o0;
  x32[(size_t)row * HID_ + 32 + lane] = o1;
  x16[(size_t)row * HID_ + lane] = (half_t)o0;
  x16[(size_t)row * HID_ + 32 + lane] = (half_t)o1;
}

// final fc: out[b, l-1, c] = x[b,l,:] @ fc_w[:,c] + fc_b[c], l in 1..1023
__global__ void k_final(const float* __restrict__ x, const float* __restrict__ fw,
                        const float* __restrict__ fb, float* __restrict__ out) {
  int i = blockIdx.x * blockDim.x + threadIdx.x;
  int total = B_ * (L_ - 1) * 6;
  if (i >= total) return;
  int c = i % 6;
  int t = i / 6;
  int l = t % (L_ - 1) + 1;
  int b = t / (L_ - 1);
  const float* xr = x + ((size_t)b * L_ + l) * HID_;
  float acc = fb[c];
#pragma unroll
  for (int d = 0; d < HID_; ++d) acc += xr[d] * fw[d * 6 + c];
  out[i] = acc;
}

// ---------------------------------------------------------------- launch

extern "C" void kernel_launch(void* const* d_in, const int* in_sizes, int n_in,
                              void* d_out, int out_size, void* d_ws,
                              size_t ws_size, hipStream_t stream) {
  (void)in_sizes; (void)n_in; (void)out_size; (void)ws_size;
  const float* features = (const float*)d_in[0];
  const float* Wq = (const float*)d_in[1];
  const float* bq = (const float*)d_in[2];
  const float* Wk = (const float*)d_in[3];
  const float* bk = (const float*)d_in[4];
  const float* Wv = (const float*)d_in[5];
  const float* bv = (const float*)d_in[6];
  const float* Wo = (const float*)d_in[7];
  const float* bo = (const float*)d_in[8];
  const float* ln1_g = (const float*)d_in[9];
  const float* ln1_b = (const float*)d_in[10];
  const float* Wi = (const float*)d_in[11];
  const float* bi = (const float*)d_in[12];
  const float* Wd = (const float*)d_in[13];
  const float* bd = (const float*)d_in[14];
  const float* ln2_g = (const float*)d_in[15];
  const float* ln2_b = (const float*)d_in[16];
  const float* rel = (const float*)d_in[17];
  const float* fcw = (const float*)d_in[18];
  const float* fcb = (const float*)d_in[19];
  float* out = (float*)d_out;

  char* p = (char*)d_ws;
  auto alloc = [&](size_t bytes) {
    char* r = p;
    p += (bytes + 255) & ~(size_t)255;
    return r;
  };
  float* x32 = (float*)alloc((size_t)BL_ * HID_ * 4);
  float* tmp = (float*)alloc((size_t)BL_ * HID_ * 4);
  half_t* x16 = (half_t*)alloc((size_t)BL_ * HID_ * 2);
  half_t* q16 = (half_t*)alloc((size_t)BL_ * HID_ * 2);
  half_t* k16 = (half_t*)alloc((size_t)BL_ * HID_ * 2);
  half_t* vT16 = (half_t*)alloc((size_t)BL_ * HID_ * 2);
  half_t* ctx16 = (half_t*)alloc((size_t)BL_ * HID_ * 2);
  half_t* inter16 = (half_t*)alloc((size_t)BL_ * INTER_ * 2);
  half_t* wq16 = (half_t*)alloc((size_t)NL_ * HID_ * HID_ * 2);
  half_t* wk16 = (half_t*)alloc((size_t)NL_ * HID_ * HID_ * 2);
  half_t* wv16 = (half_t*)alloc((size_t)NL_ * HID_ * HID_ * 2);
  half_t* wo16 = (half_t*)alloc((size_t)NL_ * HID_ * HID_ * 2);
  half_t* wi16 = (half_t*)alloc((size_t)NL_ * HID_ * INTER_ * 2);
  half_t* wd16 = (half_t*)alloc((size_t)NL_ * INTER_ * HID_ * 2);
  half_t* rel16 = (half_t*)alloc((size_t)NL_ * (2 * MAXP_ - 1) * HD_ * 2);

  const int nw = NL_ * HID_ * HID_;              // 16384
  const int nwi = NL_ * HID_ * INTER_;           // 8192
  const int nrel = NL_ * (2 * MAXP_ - 1) * HD_;  // 131008

  k_swizzle_w<<<(nw + 255) / 256, 256, 0, stream>>>(Wq, wq16, HID_, HID_, nw);
  k_swizzle_w<<<(nw + 255) / 256, 256, 0, stream>>>(Wk, wk16, HID_, HID_, nw);
  k_swizzle_w<<<(nw + 255) / 256, 256, 0, stream>>>(Wv, wv16, HID_, HID_, nw);
  k_swizzle_w<<<(nw + 255) / 256, 256, 0, stream>>>(Wo, wo16, HID_, HID_, nw);
  k_swizzle_w<<<(nwi + 255) / 256, 256, 0, stream>>>(Wi, wi16, HID_, INTER_, nwi);
  k_swizzle_w<<<(nwi + 255) / 256, 256, 0, stream>>>(Wd, wd16, INTER_, HID_, nwi);
  k_cast_f32_f16<<<(nrel + 255) / 256, 256, 0, stream>>>(rel, rel16, nrel);
  k_init_x<<<(BL_ * HID_ + 255) / 256, 256, 0, stream>>>(features, x32, x16,
                                                         BL_ * HID_);

  for (int i = 0; i < NL_; ++i) {
    k_gemm_qkv<<<dim3(BL_ / 16, 12), 32, 0, stream>>>(
        x16, wq16 + i * HID_ * HID_, wk16 + i * HID_ * HID_,
        wv16 + i * HID_ * HID_, bq + i * HID_, bk + i * HID_, bv + i * HID_,
        q16, k16, vT16);
    k_attention<<<dim3(L_ / 16, B_ * NH_), 32, 0, stream>>>(
        q16, k16, vT16, rel16 + (size_t)i * (2 * MAXP_ - 1) * HD_, ctx16);
    k_proj_res<<<dim3(BL_ / 16, 4), 32, 0, stream>>>(
        ctx16, wo16 + i * HID_ * HID_, bo + i * HID_, x32, tmp);
    k_layernorm<<<BL_, 32, 0, stream>>>(tmp, ln1_g + i * HID_,
                                        ln1_b + i * HID_, x32, x16);
    k_ffn_up<<<dim3(BL_ / 16, 2), 32, 0, stream>>>(
        x16, wi16 + i * HID_ * INTER_, bi + i * INTER_, inter16);
    k_ffn_down<<<dim3(BL_ / 16, 4), 32, 0, stream>>>(
        inter16, wd16 + i * INTER_ * HID_, bd + i * HID_, x32, tmp);
    k_layernorm<<<BL_, 32, 0, stream>>>(tmp, ln2_g + i * HID_,
                                        ln2_b + i * HID_, x32, x16);
  }
  k_final<<<(B_ * (L_ - 1) * 6 + 255) / 256, 256, 0, stream>>>(x32, fcw, fcb,
                                                               out);
}
